// GNNModel_15358803051045
// MI455X (gfx1250) — matrix operand order repaired and checked by reading, not compiled, not run on hardware
//
#include <hip/hip_runtime.h>
#include <math.h>

typedef _Float16 half_t;
typedef __attribute__((ext_vector_type(16))) _Float16 v16h;
typedef __attribute__((ext_vector_type(8)))  float    v8f;

#define HID 64
#define LDSPAD 80   // halfs per row: 160B row stride, 16B aligned

// float atomic max via integer ordering trick (init must be -inf)
__device__ __forceinline__ void atomicMaxF(float* addr, float val) {
    if (val >= 0.0f) atomicMax((int*)addr, __float_as_int(val));
    else             atomicMin((unsigned int*)addr, __float_as_uint(val));
}

// Hout[n][j] = sum_k act(A[n][k]) * W[k][j]   for FULL 16-row tiles only.
// act(x) = fuse ? tanh(x + bias[k]) : x
// One wave computes a 16x64 tile via 8x v_wmma_f32_16x16x32_f16.
// No bounds checks: host guarantees this kernel only covers rows < (N/16)*16.
__global__ __launch_bounds__(256) void gat_gemm_wmma(
    const float* __restrict__ A, const float* __restrict__ W,
    const float* __restrict__ bias, int fuse,
    float* __restrict__ Hout, int numTiles)
{
    __shared__ __align__(16) half_t Wl[HID][LDSPAD];  // [col][k], transposed W
    int tid = threadIdx.x;
    for (int i = tid; i < HID * HID; i += 256) {
        int k = i >> 6, n = i & 63;
        Wl[n][k] = (half_t)W[i];        // W is [k][n] row-major
    }
    __syncthreads();

    int lane = tid & 31;
    // force scalar: tile id is wave-uniform -> scalar branch, scalar addressing
    int wave = __builtin_amdgcn_readfirstlane(tid >> 5);
    int tile = blockIdx.x * 8 + wave;
    if (tile >= numTiles) return;       // scalar branch: EXEC stays all-1 for WMMA

    // ---- A fragments (16x32 f16 per k-step), layout per ISA 7.12.2 ----
    // lane l holds row M=l&15; K runs: [khi..khi+7] and [khi+16..khi+23], khi=(l>=16)*8
    int rowA = tile * 16 + (lane & 15);
    int khi  = (lane >> 4) << 3;
    v16h afrag[2];
    for (int kt = 0; kt < 2; ++kt) {
        const float* p = A + (size_t)rowA * HID + kt * 32 + khi;
        v16h a;
        int kbase = kt * 32 + khi;
        for (int i = 0; i < 8; ++i) {
            float x0 = p[i];
            float x1 = p[16 + i];
            if (fuse) {
                x0 = tanhf(x0 + bias[kbase + i]);
                x1 = tanhf(x1 + bias[kbase + 16 + i]);
            }
            a[i]     = (half_t)x0;
            a[8 + i] = (half_t)x1;
        }
        afrag[kt] = a;
    }

    // ---- loop over 4 output column tiles ----
    int colq = lane & 15;
    int kh2  = (lane >> 4) << 3;        // B: same K order as A: [kh2..kh2+7] and [kh2+16..kh2+23]
    int rbase = tile * 16 + ((lane >> 4) << 3);  // C/D: VGPR v -> M = v + (l>=16)*8
    for (int nt = 0; nt < 4; ++nt) {
        v8f acc = {};
        for (int kt = 0; kt < 2; ++kt) {
            int k0 = kt * 32 + kh2;
            v16h b;
            for (int i = 0; i < 16; ++i) b[i] = Wl[nt * 16 + colq][k0 + i + ((i >> 3) << 3)];
            acc = __builtin_amdgcn_wmma_f32_16x16x32_f16(
                false, afrag[kt], false, b, (short)0, acc, false, false);
        }
        float* po = Hout + (size_t)rbase * HID + nt * 16 + colq;
        #pragma unroll
        for (int v = 0; v < 8; ++v) po[(size_t)v * HID] = acc[v];
    }
}

// Scalar tail for rows [rowStart, N) when N % 16 != 0 (never launched when
// N is a multiple of 16, so it costs nothing for this workload).
__global__ void gat_gemm_tail(
    const float* __restrict__ A, const float* __restrict__ W,
    const float* __restrict__ bias, int fuse,
    float* __restrict__ Hout, int rowStart, int N)
{
    int idx = blockIdx.x * blockDim.x + threadIdx.x;
    int nElem = (N - rowStart) * HID;
    if (idx >= nElem) return;
    int r = rowStart + idx / HID;
    int j = idx % HID;
    float acc = 0.0f;
    for (int k = 0; k < HID; ++k) {
        float xv = A[(size_t)r * HID + k];
        if (fuse) xv = tanhf(xv + bias[k]);
        acc += xv * W[k * HID + j];
    }
    Hout[(size_t)r * HID + j] = acc;
}

// s_src[n] = h[n,:].a_src ; s_dst[n] = h[n,:].a_dst  (one wave per node)
__global__ __launch_bounds__(256) void gat_scores(
    const float* __restrict__ H, const float* __restrict__ a_src,
    const float* __restrict__ a_dst,
    float* __restrict__ s_src, float* __restrict__ s_dst, int N)
{
    int lane  = threadIdx.x & 31;
    int wid   = (int)((blockIdx.x * blockDim.x + threadIdx.x) >> 5);
    int nwav  = (int)((gridDim.x * blockDim.x) >> 5);
    float as0 = a_src[lane], as1 = a_src[lane + 32];
    float ad0 = a_dst[lane], ad1 = a_dst[lane + 32];
    for (int n = wid; n < N; n += nwav) {
        float v0 = H[(size_t)n * HID + lane];
        float v1 = H[(size_t)n * HID + 32 + lane];
        float ps = v0 * as0 + v1 * as1;
        float pd = v0 * ad0 + v1 * ad1;
        for (int off = 16; off > 0; off >>= 1) {
            ps += __shfl_xor(ps, off, 32);
            pd += __shfl_xor(pd, off, 32);
        }
        if (lane == 0) { s_src[n] = ps; s_dst[n] = pd; }
    }
}

__global__ void gat_init(float* __restrict__ m, float* __restrict__ denom,
                         float4* __restrict__ agg4, int N)
{
    int i = blockIdx.x * blockDim.x + threadIdx.x;
    if (i < N * (HID / 4)) agg4[i] = make_float4(0.f, 0.f, 0.f, 0.f);
    if (i < N) { m[i] = __uint_as_float(0xFF800000u); denom[i] = 0.0f; }
}

__global__ void gat_edge_max(const int* __restrict__ ei, int E, int ETOT,
                             const float* __restrict__ s_src,
                             const float* __restrict__ s_dst,
                             float* __restrict__ ebuf, float* __restrict__ m)
{
    int i = blockIdx.x * blockDim.x + threadIdx.x;
    if (i >= ETOT) return;
    int s = (i < E) ? ei[i]     : (i - E);   // self-loops appended
    int d = (i < E) ? ei[E + i] : (i - E);
    float e = s_src[s] + s_dst[d];
    e = (e >= 0.0f) ? e : 0.2f * e;          // leaky_relu(0.2)
    ebuf[i] = e;
    atomicMaxF(&m[d], e);
}

__global__ void gat_edge_exp(const int* __restrict__ ei, int E, int ETOT,
                             float* __restrict__ ebuf,
                             const float* __restrict__ m, float* __restrict__ denom)
{
    int i = blockIdx.x * blockDim.x + threadIdx.x;
    if (i >= ETOT) return;
    int d = (i < E) ? ei[E + i] : (i - E);
    float ex = expf(ebuf[i] - m[d]);
    ebuf[i] = ex;
    atomicAdd(&denom[d], ex);
}

// agg[dst,:] += alpha * H[src,:]  (one wave per edge, 2 features/lane)
// Edge index is wave-uniform: force it scalar so ei/ebuf/denom become s_loads
// and the vector pipe only carries the coalesced h reads + atomics.
__global__ __launch_bounds__(256) void gat_edge_agg(
    const int* __restrict__ ei, int E, int ETOT,
    const float* __restrict__ ebuf, const float* __restrict__ denom,
    const float* __restrict__ H, float* __restrict__ agg)
{
    long long gid = (long long)blockIdx.x * blockDim.x + threadIdx.x;
    int i = __builtin_amdgcn_readfirstlane((int)(gid >> 5));
    if (i >= ETOT) return;              // scalar branch
    int lane = threadIdx.x & 31;
    int s, d;
    if (i < E) { s = ei[i]; d = ei[E + i]; }
    else       { s = i - E; d = s; }
    float alpha = ebuf[i] / (denom[d] + 1e-16f);
    float h0 = H[(size_t)s * HID + lane];
    float h1 = H[(size_t)s * HID + 32 + lane];
    atomicAdd(&agg[(size_t)d * HID + lane],      alpha * h0);
    atomicAdd(&agg[(size_t)d * HID + 32 + lane], alpha * h1);
}

// out[n] = (agg[n,:] + b2) . Wl + bl
__global__ __launch_bounds__(256) void gat_final(
    const float* __restrict__ agg, const float* __restrict__ b2,
    const float* __restrict__ Wl, const float* __restrict__ bl,
    float* __restrict__ out, int N)
{
    int lane = threadIdx.x & 31;
    int wid  = (int)((blockIdx.x * blockDim.x + threadIdx.x) >> 5);
    int nwav = (int)((gridDim.x * blockDim.x) >> 5);
    float w0 = Wl[lane], w1 = Wl[lane + 32];
    float c0 = b2[lane], c1 = b2[lane + 32];
    for (int n = wid; n < N; n += nwav) {
        float p = (agg[(size_t)n * HID + lane] + c0) * w0 +
                  (agg[(size_t)n * HID + 32 + lane] + c1) * w1;
        for (int off = 16; off > 0; off >>= 1) p += __shfl_xor(p, off, 32);
        if (lane == 0) out[n] = p + bl[0];
    }
}

extern "C" void kernel_launch(void* const* d_in, const int* in_sizes, int n_in,
                              void* d_out, int out_size, void* d_ws, size_t ws_size,
                              hipStream_t stream) {
    (void)n_in; (void)out_size; (void)ws_size;
    const float* x   = (const float*)d_in[0];
    const int*   ei  = (const int*)  d_in[1];
    const float* W1  = (const float*)d_in[2];
    const float* as1 = (const float*)d_in[3];
    const float* ad1 = (const float*)d_in[4];
    const float* b1  = (const float*)d_in[5];
    const float* W2  = (const float*)d_in[6];
    const float* as2 = (const float*)d_in[7];
    const float* ad2 = (const float*)d_in[8];
    const float* b2  = (const float*)d_in[9];
    const float* Wlp = (const float*)d_in[10];
    const float* blp = (const float*)d_in[11];
    float* out = (float*)d_out;

    int N = in_sizes[0] / HID;
    int E = in_sizes[1] / 2;
    int ETOT = E + N;                 // with self-loops
    int numTiles = N / 16;            // full WMMA tiles
    int tailRows = N - numTiles * 16; // 0 for this workload

    char* ws = (char*)d_ws;
    size_t off = 0;
    float* bufA  = (float*)(ws + off); off += (size_t)N * HID * 4;  // h
    float* bufB  = (float*)(ws + off); off += (size_t)N * HID * 4;  // agg / next x
    float* ssrc  = (float*)(ws + off); off += (size_t)N * 4;
    float* sdst  = (float*)(ws + off); off += (size_t)N * 4;
    float* mbuf  = (float*)(ws + off); off += (size_t)N * 4;
    float* denom = (float*)(ws + off); off += (size_t)N * 4;
    float* ebuf  = (float*)(ws + off); off += (size_t)ETOT * 4;

    const int tpb = 256;
    int gInit = (N * (HID / 4) + tpb - 1) / tpb;
    int gGemm = (numTiles + 7) / 8;
    int gTail = (tailRows * HID + tpb - 1) / tpb;
    int gEdge = (ETOT + tpb - 1) / tpb;
    long long tAgg = (long long)ETOT * 32;
    int gAgg  = (int)((tAgg + tpb - 1) / tpb);
    int gWave = 1024;

    // ---------------- layer 1 ----------------
    gat_init     <<<gInit, tpb, 0, stream>>>(mbuf, denom, (float4*)bufB, N);
    gat_gemm_wmma<<<gGemm, tpb, 0, stream>>>(x, W1, nullptr, 0, bufA, numTiles);
    if (tailRows)
        gat_gemm_tail<<<gTail, tpb, 0, stream>>>(x, W1, nullptr, 0, bufA,
                                                 numTiles * 16, N);
    gat_scores   <<<gWave, tpb, 0, stream>>>(bufA, as1, ad1, ssrc, sdst, N);
    gat_edge_max <<<gEdge, tpb, 0, stream>>>(ei, E, ETOT, ssrc, sdst, ebuf, mbuf);
    gat_edge_exp <<<gEdge, tpb, 0, stream>>>(ei, E, ETOT, ebuf, mbuf, denom);
    gat_edge_agg <<<gAgg,  tpb, 0, stream>>>(ei, E, ETOT, ebuf, denom, bufA, bufB);

    // ---------------- layer 2 ----------------
    // GEMM2 reads bufB with fused tanh(x + b1); writes h2 into bufA
    gat_gemm_wmma<<<gGemm, tpb, 0, stream>>>(bufB, W2, b1, 1, bufA, numTiles);
    if (tailRows)
        gat_gemm_tail<<<gTail, tpb, 0, stream>>>(bufB, W2, b1, 1, bufA,
                                                 numTiles * 16, N);
    gat_init     <<<gInit, tpb, 0, stream>>>(mbuf, denom, (float4*)bufB, N);
    gat_scores   <<<gWave, tpb, 0, stream>>>(bufA, as2, ad2, ssrc, sdst, N);
    gat_edge_max <<<gEdge, tpb, 0, stream>>>(ei, E, ETOT, ssrc, sdst, ebuf, mbuf);
    gat_edge_exp <<<gEdge, tpb, 0, stream>>>(ei, E, ETOT, ebuf, mbuf, denom);
    gat_edge_agg <<<gAgg,  tpb, 0, stream>>>(ei, E, ETOT, ebuf, denom, bufA, bufB);

    // ---------------- readout ----------------
    gat_final    <<<gWave, tpb, 0, stream>>>(bufB, b2, Wlp, blp, out, N);
}
